// SparseConvs_36627481100426
// MI455X (gfx1250) — compile-verified, matrix operand order checked by reading, not run on hardware
//
#include <hip/hip_runtime.h>
#include <hip/hip_bf16.h>
#include <cstddef>

// ---------------------------------------------------------------------------
// MI455X (gfx1250) dense-evaluated sparse-conv backbone.
// Implicit-GEMM conv via v_wmma_f32_16x16x32_f16 (f16 in, f32 acc).
//  - Activation patch staged once in LDS (f16), reused by 16 M positions x taps.
//  - k -> LDS-offset LUT in LDS: division-free, branch-free inner loop.
//  - Weights pre-converted to f16, zero-padded to (NT*16, K32): B fragments are
//    two aligned 128-bit vector loads (L2-resident, 192MB L2 holds all weights).
//  - One wave computes a 16(x) x 16*NT(cout) tile: A fragment reused by NT WMMAs.
//  - Dual accumulators (even/odd K chunks) break the WMMA C->D RAW chain so
//    consecutive v_wmma ops pipeline on the XDL instead of stalling 5 slots.
//  - BN + ReLU + mask fused in epilogue; sp-layers also emit downsampled mask.
// ---------------------------------------------------------------------------

typedef __attribute__((ext_vector_type(16))) _Float16       v16h;
typedef __attribute__((ext_vector_type(8)))  _Float16       v8h;
typedef __attribute__((ext_vector_type(8)))  float          v8f;
typedef __attribute__((ext_vector_type(8)))  unsigned short v8us;

#define BN_EPS 1e-3f
#define LDS_A_HALVES 9600   // max patch rows*PW = 288*33 = 9504, + 32 zero pad
#define ZPAD (LDS_A_HALVES - 32)
#define KMAX32 1152         // max K32 over all layers (128*9)

struct CP {
    const float*    in;        // (cin, Zi, Yi, Xi)
    const _Float16* wh;        // (NT*16, K32) f16, zero-padded, flip pre-applied
    const float*    g;         // BN gamma
    const float*    bb;        // BN beta
    const float*    mn;        // BN mean
    const float*    vr;        // BN var
    const float*    mask_mul;  // (Zo,Yo,Xo)  used when mask_out == nullptr (subm)
    const float*    mask_in;   // (Zi,Yi,Xi)  used when mask_out != nullptr (sp)
    float*          out;       // (cout, Zo, Yo, Xo)
    float*          mask_out;  // (Zo,Yo,Xo) or nullptr
    int cin, cout;
    int Zi, Yi, Xi;
    int Zo, Yo, Xo;
    int kz, ky, kx;
    int sz, sy, sx;
    int pz, py, px;            // pad_lo per dim
    int dil;                   // input (lhs) dilation: 1 normal, 2 transposed conv
    int Ktot;                  // cin*kz*ky*kx
    int K32;                   // Ktot rounded up to 32
    int PW;                    // patch row width = 15*sx + kx
};

// Read input at dilated-space coords (zt,yt,xt); OOB / dilation holes -> 0.
__device__ __forceinline__ float fetch_in(const float* in, int c, int zt, int yt, int xt,
                                          int Zi, int Yi, int Xi, int dil) {
    const int Zd = (Zi - 1) * dil + 1;
    const int Yd = (Yi - 1) * dil + 1;
    const int Xd = (Xi - 1) * dil + 1;
    if ((unsigned)zt >= (unsigned)Zd) return 0.f;
    if ((unsigned)yt >= (unsigned)Yd) return 0.f;
    if ((unsigned)xt >= (unsigned)Xd) return 0.f;
    if (dil == 2) {
        if ((zt | yt | xt) & 1) return 0.f;
        zt >>= 1; yt >>= 1; xt >>= 1;
    }
    return in[(((size_t)c * Zi + zt) * Yi + yt) * (size_t)Xi + xt];
}

union UAcc { v8f v; float f[8]; };

template <int NT>
__global__ __launch_bounds__(32)
void conv_wmma_kernel(CP p) {
    __shared__ __align__(16) _Float16       Alds[LDS_A_HALVES];
    __shared__ __align__(16) unsigned short kofftab[KMAX32];
    __shared__ float smask[16];

    const int lane = threadIdx.x;
    const int yo   = blockIdx.y;
    const int zo   = blockIdx.z;
    const int x0   = (int)blockIdx.x << 4;

    const int kyx  = p.ky * p.kx;
    const int kzyx = p.kz * kyx;
    const int PW   = p.PW;

    // ---- Zero pad region (targets of out-of-range k in the LUT) ------------
    Alds[ZPAD + lane] = (_Float16)0.f;

    // ---- Build k -> LDS-offset LUT (divisions amortized: K32/32 iters) -----
    for (int k = lane; k < p.K32; k += 32) {
        unsigned short off = (unsigned short)ZPAD;
        if (k < p.Ktot) {
            const int c  = k / kzyx;
            int rem      = k - c * kzyx;
            const int dz = rem / kyx;  rem -= dz * kyx;
            const int dy = rem / p.kx;
            const int dx = rem - dy * p.kx;
            off = (unsigned short)(((c * p.kz + dz) * p.ky + dy) * PW + dx);
        }
        kofftab[k] = off;
    }

    // ---- Stage activation receptive-field patch into LDS as f16 ------------
    // Division-free: maintain (c,dz,dy,j) incrementally while idx += 32.
    {
        const int rows  = p.cin * p.kz * p.ky;
        const int total = rows * PW;
        int j = lane, c = 0, dz = 0, dy = 0;
        if (j >= PW) {  // lane < 32, PW >= 16 -> at most one row advance
            j -= PW;
            if (++dy == p.ky) { dy = 0; if (++dz == p.kz) { dz = 0; ++c; } }
        }
        for (int idx = lane; idx < total; idx += 32) {
            const int zt = zo * p.sz + dz - p.pz;
            const int yt = yo * p.sy + dy - p.py;
            const int xd = x0 * p.sx - p.px + j;
            Alds[idx] = (_Float16)fetch_in(p.in, c, zt, yt, xd, p.Zi, p.Yi, p.Xi, p.dil);
            j += 32;
            while (j >= PW) {
                j -= PW;
                if (++dy == p.ky) { dy = 0; if (++dz == p.kz) { dz = 0; ++c; } }
            }
        }
    }

    // ---- Mask for the 16 output x positions (compute for sp, read for subm)
    if (lane < 16) {
        const int x = x0 + lane;
        float mv = 0.f;
        if (x < p.Xo) {
            if (p.mask_out) {  // sp layer: any occupied in window (dil==1 here)
                for (int dz = 0; dz < p.kz && mv == 0.f; ++dz)
                    for (int dy = 0; dy < p.ky && mv == 0.f; ++dy)
                        for (int dx = 0; dx < p.kx; ++dx) {
                            const int zt = zo * p.sz + dz - p.pz;
                            const int yt = yo * p.sy + dy - p.py;
                            const int xi = x  * p.sx + dx - p.px;
                            if ((unsigned)zt < (unsigned)p.Zi &&
                                (unsigned)yt < (unsigned)p.Yi &&
                                (unsigned)xi < (unsigned)p.Xi) {
                                if (p.mask_in[((size_t)zt * p.Yi + yt) * (size_t)p.Xi + xi] != 0.f) {
                                    mv = 1.f; break;
                                }
                            }
                        }
                p.mask_out[((size_t)zo * p.Yo + yo) * (size_t)p.Xo + x] = mv;
            } else {
                mv = p.mask_mul[((size_t)zo * p.Yo + yo) * (size_t)p.Xo + x];
            }
        }
        smask[lane] = mv;
    }
    __syncthreads();

    // ---- Implicit GEMM main loop: branch/division-free ---------------------
    const int mrow = lane & 15;     // M row (A) / N column (B) for this lane
    const int lh   = lane >> 4;     // lane-half selects K sub-block
    const int msx  = mrow * p.sx;

    const _Float16* wcol[NT];
#pragma unroll
    for (int t = 0; t < NT; ++t) {
        wcol[t] = p.wh + (size_t)(t * 16 + mrow) * p.K32;
        __builtin_prefetch(wcol[t], 0, 3);   // global_prefetch of weight column
    }

    UAcc accE[NT], accO[NT];
    const v8f zero = {0.f, 0.f, 0.f, 0.f, 0.f, 0.f, 0.f, 0.f};
#pragma unroll
    for (int t = 0; t < NT; ++t) { accE[t].v = zero; accO[t].v = zero; }

    // One K-chunk of 32 into the given accumulator set.
    auto chunk = [&](int kc, UAcc* a) {
        // 16-bit fragment layout: k = K0 + 16*(e>>3) + 8*(lane>=16) + (e&7)
        const int kb   = (kc << 5) + (lh << 3);
        const v8us ko0 = *(const v8us*)&kofftab[kb];
        const v8us ko1 = *(const v8us*)&kofftab[kb + 16];

        union { v16h v; _Float16 h[16]; } af;
#pragma unroll
        for (int e = 0; e < 8; ++e) af.h[e]     = Alds[(int)ko0[e] + msx];
#pragma unroll
        for (int e = 0; e < 8; ++e) af.h[e + 8] = Alds[(int)ko1[e] + msx];

#pragma unroll
        for (int t = 0; t < NT; ++t) {
            union { v16h v; v8h p2[2]; } bf;
            bf.p2[0] = *(const v8h*)(wcol[t] + kb);
            bf.p2[1] = *(const v8h*)(wcol[t] + kb + 16);
            a[t].v = __builtin_amdgcn_wmma_f32_16x16x32_f16(
                false, af.v, false, bf.v, (short)0, a[t].v, false, false);
        }
    };

    const int nchunks = p.K32 >> 5;
    int kc = 0;
    for (; kc + 1 < nchunks; kc += 2) {   // even -> accE, odd -> accO: no RAW chain
        chunk(kc,     accE);
        chunk(kc + 1, accO);
    }
    if (kc < nchunks) chunk(kc, accE);    // tail chunk

    // ---- Fused BN + ReLU + mask epilogue -----------------------------------
#pragma unroll
    for (int t = 0; t < NT; ++t) {
        const int n = t * 16 + mrow;
        if (n < p.cout) {
            const float inv  = p.g[n] * rsqrtf(p.vr[n] + BN_EPS);
            const float bias = p.bb[n] - p.mn[n] * inv;
            const size_t base = ((size_t)n * p.Zo + zo) * (size_t)p.Yo * p.Xo
                              + (size_t)yo * p.Xo;
            UAcc acc;
            acc.v = accE[t].v + accO[t].v;
#pragma unroll
            for (int r = 0; r < 8; ++r) {
                const int mm = r + (lh << 3);   // C/D layout: M = r + 8*(lane>=16)
                const int x  = x0 + mm;
                if (x < p.Xo) {
                    float y = acc.f[r] * inv + bias;
                    y = y > 0.f ? y : 0.f;
                    p.out[base + x] = y * smask[mm];
                }
            }
        }
    }
}

// ---- f32 -> f16 weight pre-convert; zero-pads (rows16 x K32); optional flip -
__global__ void wconv_kernel(const float* __restrict__ w, _Float16* __restrict__ wh,
                             int cout, int cin, int kz, int ky, int kx,
                             int K, int K32, int rows16, int flip) {
    const int i = blockIdx.x * blockDim.x + threadIdx.x;
    if (i >= rows16 * K32) return;
    const int n = i / K32;
    const int k = i - n * K32;
    _Float16 val = (_Float16)0.f;
    if (n < cout && k < K) {
        int ks = k;
        if (flip) {
            const int kyx = ky * kx, kzyx = kz * kyx;
            const int c  = k / kzyx;
            int rem      = k - c * kzyx;
            const int dz = rem / kyx;  rem -= dz * kyx;
            const int dy = rem / kx;
            const int dx = rem - dy * kx;
            ks = ((c * kz + (kz - 1 - dz)) * ky + (ky - 1 - dy)) * kx + (kx - 1 - dx);
        }
        val = (_Float16)w[(size_t)n * K + ks];
    }
    wh[i] = val;
}

// ---- Scatter voxels into dense grid + occupancy mask ------------------------
__global__ void scatter_kernel(const float* __restrict__ vf, const int* __restrict__ coords,
                               float* __restrict__ dense, float* __restrict__ mask,
                               int nv, int Y, int X, size_t FULL) {
    const int i = blockIdx.x * blockDim.x + threadIdx.x;
    if (i >= nv) return;
    const int z = coords[4 * i + 1];
    const int y = coords[4 * i + 2];
    const int x = coords[4 * i + 3];
    const size_t lin = ((size_t)z * Y + y) * (size_t)X + x;
#pragma unroll
    for (int c = 0; c < 4; ++c) dense[(size_t)c * FULL + lin] = vf[4 * i + c];
    mask[lin] = 1.f;
}

// ---- m2d = any(|d2| > 0) over channels --------------------------------------
__global__ void mask2d_kernel(const float* __restrict__ d2, float* __restrict__ m2d,
                              int HW, int C) {
    const int i = blockIdx.x * blockDim.x + threadIdx.x;
    if (i >= HW) return;
    float mv = 0.f;
    for (int c = 0; c < C; ++c) {
        if (fabsf(d2[(size_t)c * HW + i]) > 0.f) { mv = 1.f; break; }
    }
    m2d[i] = mv;
}

// ---------------------------------------------------------------------------
struct Layer { const float *w, *g, *b, *m, *v; };

static void run_conv(hipStream_t s, const Layer& l,
                     const float* in, float* out,
                     const float* mask_mul, const float* mask_in, float* mask_out,
                     _Float16* wh,
                     int cin, int cout,
                     int Zi, int Yi, int Xi, int Zo, int Yo, int Xo,
                     int kz, int ky, int kx, int sz, int sy, int sx,
                     int pz, int py, int px, int dil, int flip) {
    const int K      = cin * kz * ky * kx;
    const int K32    = (K + 31) & ~31;
    const int NT     = (cout + 15) / 16;
    const int rows16 = NT * 16;
    const int tw     = rows16 * K32;
    wconv_kernel<<<(tw + 255) / 256, 256, 0, s>>>(l.w, wh, cout, cin, kz, ky, kx,
                                                  K, K32, rows16, flip);

    CP p;
    p.in = in; p.wh = wh;
    p.g = l.g; p.bb = l.b; p.mn = l.m; p.vr = l.v;
    p.mask_mul = mask_mul; p.mask_in = mask_in;
    p.out = out; p.mask_out = mask_out;
    p.cin = cin; p.cout = cout;
    p.Zi = Zi; p.Yi = Yi; p.Xi = Xi;
    p.Zo = Zo; p.Yo = Yo; p.Xo = Xo;
    p.kz = kz; p.ky = ky; p.kx = kx;
    p.sz = sz; p.sy = sy; p.sx = sx;
    p.pz = pz; p.py = py; p.px = px;
    p.dil = dil;
    p.Ktot = K;
    p.K32 = K32;
    p.PW = 15 * sx + kx;

    dim3 grid((Xo + 15) / 16, Yo, Zo);
    switch (NT) {
        case 1:  conv_wmma_kernel<1><<<grid, 32, 0, s>>>(p); break;
        case 2:  conv_wmma_kernel<2><<<grid, 32, 0, s>>>(p); break;
        default: conv_wmma_kernel<4><<<grid, 32, 0, s>>>(p); break;
    }
}

extern "C" void kernel_launch(void* const* d_in, const int* in_sizes, int n_in,
                              void* d_out, int out_size, void* d_ws, size_t ws_size,
                              hipStream_t stream) {
    (void)out_size;
    if (n_in < 118) return;

    const float* vf     = (const float*)d_in[0];
    const int*   coords = (const int*)d_in[1];
    const int    NV     = in_sizes[0] / 4;

    int idx = 3;  // skip batch_size
    auto nextL = [&]() {
        Layer l{ (const float*)d_in[idx + 0], (const float*)d_in[idx + 1],
                 (const float*)d_in[idx + 2], (const float*)d_in[idx + 3],
                 (const float*)d_in[idx + 4] };
        idx += 5;
        return l;
    };
    // Order matches setup_inputs() insertion order.
    Layer c_in = nextL(), c1 = nextL();
    Layer c2a = nextL(), c2b = nextL(), c2c = nextL();
    Layer c3a = nextL(), c3b = nextL(), c3c = nextL();
    Layer c4a = nextL(), c4b = nextL(), c4c = nextL();
    Layer c5  = nextL();
    Layer m1a = nextL(), m1b = nextL(), m1c = nextL(), m1d = nextL();
    Layer m2a = nextL(), m2b = nextL(), m2c = nextL(), m2d_ = nextL();
    Layer inv1 = nextL(), inv2 = nextL(), outL = nextL();

    // ---- Workspace layout (floats) -----------------------------------------
    const size_t FULL = (size_t)41 * 400 * 352;   // 5,772,800
    float* ws  = (float*)d_ws;
    float* A   = ws;                  // 8*FULL
    float* B   = ws + 8  * FULL;      // 8*FULL (dense0 lives in first 4*FULL)
    float* MA  = ws + 16 * FULL;      // FULL
    float* MB  = ws + 17 * FULL;      // FULL
    float* T0  = ws + 18 * FULL;      // 160000 each
    float* T1  = T0 + 160000;
    float* T2  = T1 + 160000;
    float* T3  = T2 + 160000;
    float* M2D = T3 + 160000;         // 4400 each
    float* M1M = M2D + 4400;
    float* M2M = M1M + 4400;
    const size_t need_floats = 18 * FULL + 4 * 160000 + 3 * 4400;
    _Float16* WH = (_Float16*)(ws + need_floats);       // 80000 halves reserved
    const size_t need_bytes = need_floats * 4 + 80000 * 2;
    if (ws_size < need_bytes) return;

    // ---- Scatter into dense grid -------------------------------------------
    hipMemsetAsync(B,  0, 4 * FULL * sizeof(float), stream);  // dense0 (4ch)
    hipMemsetAsync(MA, 0, FULL * sizeof(float), stream);      // mask0
    scatter_kernel<<<(NV + 255) / 256, 256, 0, stream>>>(vf, coords, B, MA, NV, 400, 352, FULL);

    // ---- 3D stage -----------------------------------------------------------
    run_conv(stream, c_in, B, A, MA, nullptr, nullptr, WH, 4, 8,  41,400,352, 41,400,352, 3,3,3, 1,1,1, 1,1,1, 1,0);
    run_conv(stream, c1,   A, B, MA, nullptr, nullptr, WH, 8, 8,  41,400,352, 41,400,352, 3,3,3, 1,1,1, 1,1,1, 1,0);
    run_conv(stream, c2a,  B, A, nullptr, MA, MB,      WH, 8, 16, 41,400,352, 20,200,176, 3,3,3, 2,2,2, 0,1,1, 1,0);
    run_conv(stream, c2b,  A, B, MB, nullptr, nullptr, WH, 16,16, 20,200,176, 20,200,176, 3,3,3, 1,1,1, 1,1,1, 1,0);
    run_conv(stream, c2c,  B, A, MB, nullptr, nullptr, WH, 16,16, 20,200,176, 20,200,176, 3,3,3, 1,1,1, 1,1,1, 1,0);
    run_conv(stream, c3a,  A, B, nullptr, MB, MA,      WH, 16,32, 20,200,176, 10,100,88,  3,3,3, 2,2,2, 1,1,1, 1,0);
    run_conv(stream, c3b,  B, A, MA, nullptr, nullptr, WH, 32,32, 10,100,88,  10,100,88,  3,3,3, 1,1,1, 1,1,1, 1,0);
    run_conv(stream, c3c,  A, B, MA, nullptr, nullptr, WH, 32,32, 10,100,88,  10,100,88,  3,3,3, 1,1,1, 1,1,1, 1,0);
    run_conv(stream, c4a,  B, A, nullptr, MA, MB,      WH, 32,32, 10,100,88,  5,50,44,    3,3,3, 2,2,2, 1,1,1, 1,0);
    run_conv(stream, c4b,  A, B, MB, nullptr, nullptr, WH, 32,32, 5,50,44,    5,50,44,    3,3,3, 1,1,1, 1,1,1, 1,0);
    run_conv(stream, c4c,  B, A, MB, nullptr, nullptr, WH, 32,32, 5,50,44,    5,50,44,    3,3,3, 1,1,1, 1,1,1, 1,0);
    run_conv(stream, c5,   A, B, nullptr, MB, MA,      WH, 32,64, 5,50,44,    2,50,44,    3,1,1, 2,1,1, 0,0,0, 1,0);

    // ---- 2D stage (Z=1, kz=1) ----------------------------------------------
    mask2d_kernel<<<(2200 + 255) / 256, 256, 0, stream>>>(B, M2D, 2200, 128);
    run_conv(stream, m1a,  B,  T0, nullptr, M2D, M1M,    WH, 128,64, 1,50,44, 1,50,44, 1,3,3, 1,1,1, 0,1,1, 1,0);
    run_conv(stream, m1b,  T0, T1, M1M, nullptr, nullptr, WH, 64,64, 1,50,44, 1,50,44, 1,3,3, 1,1,1, 0,1,1, 1,0);
    run_conv(stream, m1c,  T1, T0, M1M, nullptr, nullptr, WH, 64,64, 1,50,44, 1,50,44, 1,3,3, 1,1,1, 0,1,1, 1,0);
    run_conv(stream, m1d,  T0, T1, M1M, nullptr, nullptr, WH, 64,64, 1,50,44, 1,50,44, 1,3,3, 1,1,1, 0,1,1, 1,0);
    // a = T1
    run_conv(stream, m2a,  T1, T2, nullptr, M1M, M2M,    WH, 64,64, 1,50,44, 1,25,22, 1,3,3, 1,2,2, 0,1,1, 1,0);
    run_conv(stream, m2b,  T2, T3, M2M, nullptr, nullptr, WH, 64,64, 1,25,22, 1,25,22, 1,3,3, 1,1,1, 0,1,1, 1,0);
    run_conv(stream, m2c,  T3, T2, M2M, nullptr, nullptr, WH, 64,64, 1,25,22, 1,25,22, 1,3,3, 1,1,1, 0,1,1, 1,0);
    run_conv(stream, m2d_, T2, T3, M2M, nullptr, nullptr, WH, 64,64, 1,25,22, 1,25,22, 1,3,3, 1,1,1, 0,1,1, 1,0);
    // b2 = T3
    run_conv(stream, inv1, T1, A,             M1M, nullptr, nullptr, WH, 64,64, 1,50,44, 1,50,44, 1,3,3, 1,1,1, 0,1,1, 1,0);
    run_conv(stream, inv2, T3, A + 64 * 2200, M1M, nullptr, nullptr, WH, 64,64, 1,25,22, 1,50,44, 1,3,3, 1,1,1, 0,1,1, 2,1);
    run_conv(stream, outL, A, (float*)d_out,  M1M, nullptr, nullptr, WH, 128,64, 1,50,44, 1,50,44, 1,3,3, 1,1,1, 0,1,1, 1,0);
}
